// CrossAttention_14955076125227
// MI455X (gfx1250) — compile-verified, matrix operand order checked by reading, not run on hardware
//
#include <hip/hip_runtime.h>
#include <hip/hip_bf16.h>

typedef __attribute__((ext_vector_type(16))) _Float16 v16h;
typedef __attribute__((ext_vector_type(8)))  _Float16 v8h;
typedef __attribute__((ext_vector_type(8)))  float    v8f;
typedef int v4i_vs __attribute__((vector_size(4 * sizeof(int))));

#define H_HEADS 8

#if defined(__has_builtin)
#  if __has_builtin(__builtin_amdgcn_global_load_async_to_lds_b128)
#    define HAVE_ASYNC_LDS 1
#  endif
#endif
#ifndef HAVE_ASYNC_LDS
#  define HAVE_ASYNC_LDS 0
#endif

// 16-byte global -> LDS copy; async (ASYNCcnt-tracked) when available.
__device__ __forceinline__ void async_copy16(const _Float16* g, _Float16* l) {
#if HAVE_ASYNC_LDS
    __builtin_amdgcn_global_load_async_to_lds_b128(
        (__attribute__((address_space(1))) v4i_vs*)(g),
        (__attribute__((address_space(3))) v4i_vs*)(l), 0, 0);
#else
    *(v8h*)l = *(const v8h*)g;
#endif
}

template <int N>
__device__ __forceinline__ void wait_async() {
#if HAVE_ASYNC_LDS && __has_builtin(__builtin_amdgcn_s_wait_asynccnt)
    __builtin_amdgcn_s_wait_asynccnt(N);
#else
    asm volatile("s_wait_asynccnt %0" ::"i"(N) : "memory");
#endif
}

// Load a 16x32 f16 WMMA A/B fragment from a row-major matrix.
// Per ISA 16-bit A layout: lane<16 -> row=lane, K={0..7,16..23};
// lane>=16 -> row=lane-16, K={8..15,24..31}. Two contiguous b128 loads.
__device__ __forceinline__ v16h load_frag(const _Float16* __restrict__ base, int stride) {
    int lane = threadIdx.x & 31;
    const _Float16* p = base + (size_t)(lane & 15) * stride + ((lane >> 4) << 3);
    v8h lo = *(const v8h*)(p);
    v8h hi = *(const v8h*)(p + 16);
    return __builtin_shufflevector(lo, hi, 0,1,2,3,4,5,6,7,8,9,10,11,12,13,14,15);
}

__device__ __forceinline__ v8f wmma16(v16h a, v16h b, v8f c) {
    return __builtin_amdgcn_wmma_f32_16x16x32_f16(false, a, false, b, (short)0, c, false, false);
}

// ---------------- LayerNorm: one block per row, optional f16 or f32 output ----
__global__ void k_layernorm(const float* __restrict__ in, const float* __restrict__ g,
                            const float* __restrict__ b, int D, int mode,
                            _Float16* __restrict__ outH, float* __restrict__ outF) {
    int row = blockIdx.x;
    const float* x = in + (size_t)row * D;
    float s = 0.f, s2 = 0.f;
    for (int i = threadIdx.x; i < D; i += blockDim.x) { float v = x[i]; s += v; s2 += v * v; }
    for (int off = 16; off > 0; off >>= 1) {
        s  += __shfl_xor(s,  off, 32);
        s2 += __shfl_xor(s2, off, 32);
    }
    __shared__ float sh[2][8];
    int wave = threadIdx.x >> 5, lane = threadIdx.x & 31;
    if (lane == 0) { sh[0][wave] = s; sh[1][wave] = s2; }
    __syncthreads();
    if (wave == 0) {
        float a  = (lane < 8) ? sh[0][lane] : 0.f;
        float a2 = (lane < 8) ? sh[1][lane] : 0.f;
        for (int off = 4; off > 0; off >>= 1) {
            a  += __shfl_xor(a,  off, 32);
            a2 += __shfl_xor(a2, off, 32);
        }
        if (lane == 0) { sh[0][0] = a; sh[1][0] = a2; }
    }
    __syncthreads();
    float mu  = sh[0][0] / (float)D;
    float var = sh[1][0] / (float)D - mu * mu;
    float inv = rsqrtf(var + 1e-5f);
    for (int i = threadIdx.x; i < D; i += blockDim.x) {
        float y = (x[i] - mu) * inv * g[i] + b[i];
        if (mode == 0) outH[(size_t)row * D + i] = (_Float16)y;
        else           outF[(size_t)row * D + i] = y;
    }
}

// ---------------- Weight convert + transpose: W[k][n] f32 -> Wt[n][k] f16 ------
__global__ void k_w2h_t(const float* __restrict__ W, _Float16* __restrict__ Wt,
                        int K, int Ncol) {
    int idx = blockIdx.x * blockDim.x + threadIdx.x;
    if (idx >= K * Ncol) return;
    int n = idx / K, k = idx % K;
    Wt[(size_t)n * K + k] = (_Float16)W[(size_t)k * Ncol + n];
}

// ---------------- WMMA GEMM: C[row0..+16][col0..+16] = A(16xK) * Bt(16xK)^T ----
// mode 0/1: out f16 [b][h][tok][64]   mode 2: out f16 [b][h][64][tok] (V^T)
// mode 3: out f32 [row][512] + bias + residual
__global__ void __launch_bounds__(128) k_gemm(
        const _Float16* __restrict__ A, const _Float16* __restrict__ Bt,
        int Kdim, int Ntok, int mode,
        _Float16* __restrict__ outH, float* __restrict__ outF,
        const float* __restrict__ bias, const float* __restrict__ resid) {
    int lane = threadIdx.x & 31;
    int wave = threadIdx.x >> 5;
    int row0 = blockIdx.x * 16;
    int col0 = (blockIdx.y * 4 + wave) * 16;
    const _Float16* arow = A  + (size_t)row0 * Kdim;
    const _Float16* brow = Bt + (size_t)col0 * Kdim;

    v8f acc = {};
    for (int k0 = 0; k0 < Kdim; k0 += 32) {
        v16h a = load_frag(arow + k0, Kdim);
        v16h b = load_frag(brow + k0, Kdim);
        acc = wmma16(a, b, acc);
    }
    // C layout: col = col0 + (lane&15); row = row0 + 8*(lane>>4) + v
    int col   = col0 + (lane & 15);
    int rbase = row0 + ((lane >> 4) << 3);
    if (mode <= 1) {
        int h = col >> 6, d = col & 63;
        #pragma unroll
        for (int v = 0; v < 8; ++v) {
            int row = rbase + v, bb = row / Ntok, t = row % Ntok;
            outH[(((size_t)bb * H_HEADS + h) * Ntok + t) * 64 + d] = (_Float16)acc[v];
        }
    } else if (mode == 2) {
        int h = col >> 6, d = col & 63;
        #pragma unroll
        for (int v = 0; v < 8; ++v) {
            int row = rbase + v, bb = row / Ntok, t = row % Ntok;
            outH[(((size_t)bb * H_HEADS + h) * 64 + d) * Ntok + t] = (_Float16)acc[v];
        }
    } else {
        #pragma unroll
        for (int v = 0; v < 8; ++v) {
            size_t row = (size_t)(rbase + v);
            outF[row * 512 + col] = acc[v] + bias[col] + resid[row * 512 + col];
        }
    }
}

// ---------------- Flash attention: 4 waves/block share (b,h); K/V tiles are
// staged into double-buffered LDS via async global->LDS DMA (ASYNCcnt). -------
__global__ void __launch_bounds__(128) k_attention(
        const _Float16* __restrict__ qh, const _Float16* __restrict__ kh,
        const _Float16* __restrict__ vT, _Float16* __restrict__ outh,
        int Ntok, int Mtok) {
    __shared__ __align__(16) _Float16 Kt[2][32 * 64];   // K tile: 32 m-rows x 64 d
    __shared__ __align__(16) _Float16 Vt[2][64 * 32];   // V^T tile: 64 d-rows x 32 m
    __shared__ __align__(16) _Float16 Plds[4][16][32];  // per-wave P re-layout

    int tid   = threadIdx.x;
    int lane  = tid & 31;
    int wave  = tid >> 5;
    int bh    = blockIdx.y;                         // b*H + h
    int nbase = (blockIdx.x * 4 + wave) * 16;

    const _Float16* qbase = qh + ((size_t)bh * Ntok + nbase) * 64;
    const _Float16* kbase = kh + (size_t)bh * Mtok * 64;
    const _Float16* vbase = vT + (size_t)bh * 64 * Mtok;

    v16h q0 = load_frag(qbase,      64);            // K = d 0..31
    v16h q1 = load_frag(qbase + 32, 64);            // K = d 32..63

    // Cooperative tile fetch: 4 async b128 ops per thread per step (8 KB total).
    auto fetch_tiles = [&](int buf, int m0) {
        #pragma unroll
        for (int j = 0; j < 2; ++j) {
            int i = tid + 128 * j;                        // 16B chunk id, 0..255
            async_copy16(kbase + (size_t)m0 * 64 + (size_t)i * 8, &Kt[buf][i * 8]);
            int d = i >> 2, mo = (i & 3) * 8;
            async_copy16(vbase + (size_t)d * Mtok + m0 + mo, &Vt[buf][d * 32 + mo]);
        }
    };

    v8f o0 = {}, o1 = {}, o2 = {}, o3 = {};
    float mrow[8], lsum[8];
    #pragma unroll
    for (int v = 0; v < 8; ++v) { mrow[v] = -3.0e38f; lsum[v] = 0.f; }
    const float scale = 0.125f;                     // 1/sqrt(64)

    fetch_tiles(0, 0);
    for (int it = 0, m0 = 0; m0 < Mtok; m0 += 32, ++it) {
        int buf = it & 1;
        if (m0 + 32 < Mtok) { fetch_tiles(buf ^ 1, m0 + 32); wait_async<4>(); }
        else                {                                 wait_async<0>(); }
        __syncthreads();                            // tiles for this step visible

        const _Float16* kt = &Kt[buf][0];
        v16h ka0 = load_frag(kt,                64);  // m0..m0+15,  d 0..31
        v16h ka1 = load_frag(kt + 32,           64);
        v16h kb0 = load_frag(kt + 16 * 64,      64);  // m0+16..m0+31
        v16h kb1 = load_frag(kt + 16 * 64 + 32, 64);
        v8f s0 = {}, s1 = {};
        s0 = wmma16(q0, ka0, s0); s0 = wmma16(q1, ka1, s0);
        s1 = wmma16(q0, kb0, s1); s1 = wmma16(q1, kb1, s1);

        float p0[8], p1[8];
        #pragma unroll
        for (int v = 0; v < 8; ++v) {
            float a = s0[v] * scale, c = s1[v] * scale;
            float t = fmaxf(a, c);                  // same row, cols j and j+16
            for (int off = 8; off > 0; off >>= 1) t = fmaxf(t, __shfl_xor(t, off, 32));
            float mnew  = fmaxf(mrow[v], t);
            float alpha = __expf(mrow[v] - mnew);
            mrow[v] = mnew;
            float e0 = __expf(a - mnew), e1 = __expf(c - mnew);
            float rs = e0 + e1;
            for (int off = 8; off > 0; off >>= 1) rs += __shfl_xor(rs, off, 32);
            lsum[v] = lsum[v] * alpha + rs;
            o0[v] *= alpha; o1[v] *= alpha; o2[v] *= alpha; o3[v] *= alpha;
            p0[v] = e0; p1[v] = e1;
        }
        // Re-layout P (C-format -> A-operand format) through LDS
        int r = ((lane >> 4) << 3);
        int c = lane & 15;
        #pragma unroll
        for (int v = 0; v < 8; ++v) {
            Plds[wave][r + v][c]      = (_Float16)p0[v];
            Plds[wave][r + v][c + 16] = (_Float16)p1[v];
        }
        asm volatile("s_wait_dscnt 0x0" ::: "memory");
        v16h pa = load_frag(&Plds[wave][0][0], 32);

        const _Float16* vt = &Vt[buf][0];
        v16h vb0 = load_frag(vt,           32);       // d 0..15
        v16h vb1 = load_frag(vt + 16 * 32, 32);
        v16h vb2 = load_frag(vt + 32 * 32, 32);
        v16h vb3 = load_frag(vt + 48 * 32, 32);
        o0 = wmma16(pa, vb0, o0);
        o1 = wmma16(pa, vb1, o1);
        o2 = wmma16(pa, vb2, o2);
        o3 = wmma16(pa, vb3, o3);
        __syncthreads();                            // tile consumed; safe to refill
    }

    #pragma unroll
    for (int v = 0; v < 8; ++v) {
        float inv = 1.0f / lsum[v];
        o0[v] *= inv; o1[v] *= inv; o2[v] *= inv; o3[v] *= inv;
    }
    int b  = bh >> 3, h = bh & 7;
    int c  = lane & 15;
    int rb = nbase + ((lane >> 4) << 3);
    #pragma unroll
    for (int v = 0; v < 8; ++v) {
        size_t rowoff = ((size_t)b * Ntok + rb + v) * 512 + (size_t)h * 64;
        outh[rowoff + c]      = (_Float16)o0[v];
        outh[rowoff + 16 + c] = (_Float16)o1[v];
        outh[rowoff + 32 + c] = (_Float16)o2[v];
        outh[rowoff + 48 + c] = (_Float16)o3[v];
    }
}

extern "C" void kernel_launch(void* const* d_in, const int* in_sizes, int n_in,
                              void* d_out, int out_size, void* d_ws, size_t ws_size,
                              hipStream_t stream) {
    (void)in_sizes; (void)n_in; (void)out_size; (void)ws_size;
    const float* x     = (const float*)d_in[0];
    const float* cond  = (const float*)d_in[1];
    const float* lnx_g = (const float*)d_in[2];
    const float* lnx_b = (const float*)d_in[3];
    const float* lnc_g = (const float*)d_in[4];
    const float* lnc_b = (const float*)d_in[5];
    const float* Wq    = (const float*)d_in[6];
    const float* Wk    = (const float*)d_in[7];
    const float* Wv    = (const float*)d_in[8];
    const float* Wo    = (const float*)d_in[9];
    const float* bo    = (const float*)d_in[10];
    const float* lnf_g = (const float*)d_in[11];
    const float* lnf_b = (const float*)d_in[12];

    const int B = 4, N = 2048, M = 2048, Dq = 512, Dc = 768, INNER = 512;

    char* ws = (char*)d_ws;
    size_t off = 0;
    auto alloc = [&](size_t bytes) -> void* {
        void* p = ws + off;
        off = (off + bytes + 255) & ~(size_t)255;
        return p;
    };
    _Float16* xn  = (_Float16*)alloc((size_t)B * N * Dq * 2);
    _Float16* cn  = (_Float16*)alloc((size_t)B * M * Dc * 2);
    _Float16* WqT = (_Float16*)alloc((size_t)Dq * INNER * 2);
    _Float16* WkT = (_Float16*)alloc((size_t)Dc * INNER * 2);
    _Float16* WvT = (_Float16*)alloc((size_t)Dc * INNER * 2);
    _Float16* WoT = (_Float16*)alloc((size_t)INNER * Dq * 2);
    _Float16* qh  = (_Float16*)alloc((size_t)B * H_HEADS * N * 64 * 2);
    _Float16* kh  = (_Float16*)alloc((size_t)B * H_HEADS * M * 64 * 2);
    _Float16* vT  = (_Float16*)alloc((size_t)B * H_HEADS * 64 * M * 2);
    _Float16* ah  = (_Float16*)alloc((size_t)B * N * INNER * 2);
    float*    tmp = (float*)   alloc((size_t)B * N * Dq * 4);

    // 1) LayerNorms -> f16
    k_layernorm<<<B * N, 256, 0, stream>>>(x,    lnx_g, lnx_b, Dq, 0, xn, nullptr);
    k_layernorm<<<B * M, 256, 0, stream>>>(cond, lnc_g, lnc_b, Dc, 0, cn, nullptr);

    // 2) Weight convert+transpose -> f16 [n][k]
    const int TW = 256;
    k_w2h_t<<<(Dq * INNER + TW - 1) / TW, TW, 0, stream>>>(Wq, WqT, Dq, INNER);
    k_w2h_t<<<(Dc * INNER + TW - 1) / TW, TW, 0, stream>>>(Wk, WkT, Dc, INNER);
    k_w2h_t<<<(Dc * INNER + TW - 1) / TW, TW, 0, stream>>>(Wv, WvT, Dc, INNER);
    k_w2h_t<<<(INNER * Dq + TW - 1) / TW, TW, 0, stream>>>(Wo, WoT, INNER, Dq);

    // 3) Q/K/V projections (WMMA)
    dim3 gq((B * N) / 16, INNER / 64);
    k_gemm<<<gq, 128, 0, stream>>>(xn, WqT, Dq, N, 0, qh, nullptr, nullptr, nullptr);
    dim3 gk((B * M) / 16, INNER / 64);
    k_gemm<<<gk, 128, 0, stream>>>(cn, WkT, Dc, M, 1, kh, nullptr, nullptr, nullptr);
    k_gemm<<<gk, 128, 0, stream>>>(cn, WvT, Dc, M, 2, vT, nullptr, nullptr, nullptr);

    // 4) Flash attention (WMMA scores + WMMA PV, async-LDS K/V pipeline)
    dim3 ga(N / 64, B * H_HEADS);
    k_attention<<<ga, 128, 0, stream>>>(qh, kh, vT, ah, N, M);

    // 5) Output projection + bias + residual (WMMA), then final LayerNorm -> f32
    dim3 go((B * N) / 16, Dq / 64);
    k_gemm<<<go, 128, 0, stream>>>(ah, WoT, INNER, N, 3, nullptr, tmp, bo, x);
    k_layernorm<<<B * N, 256, 0, stream>>>(tmp, lnf_g, lnf_b, Dq, 1, nullptr, (float*)d_out);
}
